// FeedForwardNetwork_27685359190083
// MI455X (gfx1250) — matrix operand order repaired and checked by reading, not compiled, not run on hardware
//
#include <hip/hip_runtime.h>
#include <hip/hip_bf16.h>

typedef __attribute__((ext_vector_type(8)))  float  v8f;
typedef __attribute__((ext_vector_type(16))) __bf16 v16bf;
typedef __attribute__((ext_vector_type(8)))  __bf16 v8bf;
typedef __attribute__((ext_vector_type(4)))  __bf16 v4bf;
typedef __attribute__((ext_vector_type(4)))  int    v4i;

#define BM 128
#define BN 128
#define KT 64
#define LDK 72   // KT + 8 bf16 pad -> 144B row stride = 36 banks: frag reads hit banks 0,4,..,60 (conflict-free)

#if __has_builtin(__builtin_amdgcn_global_load_async_to_lds_b128)
#define USE_ASYNC_LDS 1
#else
#define USE_ASYNC_LDS 0
#endif

__device__ __forceinline__ void async_cp_b128(const __bf16* g, __bf16* l)
{
#if USE_ASYNC_LDS
    __builtin_amdgcn_global_load_async_to_lds_b128(
        (__attribute__((address_space(1))) v4i*)(v4i*)(void*)g,
        (__attribute__((address_space(3))) v4i*)(v4i*)(void*)l,
        0, 0);
#endif
}

__device__ __forceinline__ void wait_async0()
{
#if __has_builtin(__builtin_amdgcn_s_wait_asynccnt)
    __builtin_amdgcn_s_wait_asynccnt(0);
#else
    asm volatile("s_wait_asynccnt 0x0" ::: "memory");
#endif
}

// ---------------- prep kernels ----------------

__global__ __launch_bounds__(256)
void k_cvt_f32_to_bf16(const float* __restrict__ in, __bf16* __restrict__ out, long n)
{
    long i = ((long)blockIdx.x * 256 + threadIdx.x) * 4;
    if (i + 3 < n) {
        float4 f = *(const float4*)(in + i);
        v4bf o;
        o.x = (__bf16)f.x; o.y = (__bf16)f.y; o.z = (__bf16)f.z; o.w = (__bf16)f.w;
        *(v4bf*)(out + i) = o;
    }
}

// out[n][k] = (bf16) in[k][n];  in is K x N row-major, out is N x K row-major
__global__ __launch_bounds__(256)
void k_transpose_cvt(const float* __restrict__ in, __bf16* __restrict__ out, int K, int N)
{
    __shared__ float tile[32 * 33];
    const int tx = threadIdx.x & 31, ty = threadIdx.x >> 5;
    const int n0 = blockIdx.x * 32, k0 = blockIdx.y * 32;
    #pragma unroll
    for (int r = 0; r < 4; ++r) {
        int k = ty + r * 8;
        tile[k * 33 + tx] = in[(long)(k0 + k) * N + n0 + tx];
    }
    __syncthreads();
    #pragma unroll
    for (int r = 0; r < 4; ++r) {
        int n = ty + r * 8;
        out[(long)(n0 + n) * K + k0 + tx] = (__bf16)tile[tx * 33 + n];
    }
}

// ---------------- bf16 WMMA GEMM:  C = A[M][K] * Bt[N][K]^T ----------------
// EPI==0 : h = bf16(relu(acc + bias))  -> outB
// EPI==1 : out = mask(row) * (acc + bias) -> outF (fp32)

template<int EPI>
__global__ __launch_bounds__(256, 1)
void k_ffn_gemm(const __bf16* __restrict__ A,    // [M][K] row-major
                const __bf16* __restrict__ Bt,   // [N][K] row-major (pre-transposed weight)
                const float*  __restrict__ bias, // [N]
                const int*    __restrict__ padmask, // [M], 0 = real token (EPI==1 only)
                float*        __restrict__ outF,
                __bf16*       __restrict__ outB,
                int M, int N, int K)
{
    __shared__ __bf16 sA[2][BM * LDK];   // 2 x 18KB
    __shared__ __bf16 sB[2][BN * LDK];   // 2 x 18KB  (72KB total, WGP has 320KB)

    const int t    = threadIdx.x;
    const int lane = t & 31;
    const int wv   = t >> 5;       // 8 waves
    const int wm   = wv & 1;       // 2 along M (64 rows each)
    const int wn   = wv >> 1;      // 4 along N (32 cols each)
    const int lhi  = lane >> 4;    // half-wave select
    const int l15  = lane & 15;

    const long m0 = (long)blockIdx.y * BM;
    const long n0 = (long)blockIdx.x * BN;

    v8f acc[4][2] = {};

    // staging: 128x64 bf16 tile = 1024 x 16B chunks; 4 chunks / thread / matrix
    int rowS[4], kcS[4];
    #pragma unroll
    for (int i = 0; i < 4; ++i) {
        int c = t + 256 * i;
        rowS[i] = c >> 3;          // 8 chunks per row
        kcS[i]  = (c & 7) * 8;     // bf16 element offset
    }

    const int steps = K / KT;

#if USE_ASYNC_LDS
    // -------- async DMA global->LDS pipeline (no staging VGPRs, ASYNCcnt-tracked) --------
    #pragma unroll
    for (int i = 0; i < 4; ++i) {
        async_cp_b128(A  + (m0 + rowS[i]) * K + kcS[i], &sA[0][rowS[i] * LDK + kcS[i]]);
        async_cp_b128(Bt + (n0 + rowS[i]) * K + kcS[i], &sB[0][rowS[i] * LDK + kcS[i]]);
    }
    wait_async0();
    __syncthreads();
#else
    uint4 ra[4], rb[4];
    #pragma unroll
    for (int i = 0; i < 4; ++i) {
        ra[i] = *(const uint4*)(A  + (m0 + rowS[i]) * K + kcS[i]);
        rb[i] = *(const uint4*)(Bt + (n0 + rowS[i]) * K + kcS[i]);
    }
    #pragma unroll
    for (int i = 0; i < 4; ++i) {
        *(uint4*)(&sA[0][rowS[i] * LDK + kcS[i]]) = ra[i];
        *(uint4*)(&sB[0][rowS[i] * LDK + kcS[i]]) = rb[i];
    }
    __syncthreads();
#endif

    for (int s = 0; s < steps; ++s) {
        const int cur = s & 1;

#if USE_ASYNC_LDS
        if (s + 1 < steps) {
            const int  nxt = 1 - cur;
            const long kb  = (long)(s + 1) * KT;
            #pragma unroll
            for (int i = 0; i < 4; ++i) {
                async_cp_b128(A  + (m0 + rowS[i]) * K + kb + kcS[i], &sA[nxt][rowS[i] * LDK + kcS[i]]);
                async_cp_b128(Bt + (n0 + rowS[i]) * K + kb + kcS[i], &sB[nxt][rowS[i] * LDK + kcS[i]]);
            }
        }
#else
        uint4 pra[4], prb[4];
        if (s + 1 < steps) {
            const long kb = (long)(s + 1) * KT;
            #pragma unroll
            for (int i = 0; i < 4; ++i) {
                pra[i] = *(const uint4*)(A  + (m0 + rowS[i]) * K + kb + kcS[i]);
                prb[i] = *(const uint4*)(Bt + (n0 + rowS[i]) * K + kb + kcS[i]);
            }
        }
#endif

        const __bf16* pa = &sA[cur][0];
        const __bf16* pb = &sB[cur][0];

        #pragma unroll
        for (int kk = 0; kk < KT; kk += 32) {
            // ---- fragment loads per CDNA5 16-bit WMMA VGPR layouts ----
            v16bf af[4], bfr[2];
            #pragma unroll
            for (int ni = 0; ni < 2; ++ni) {
                // B 32x16: lane<16 -> K {0..7,16..23}; lane>=16 -> K {8..15,24..31}; col = lane&15
                const __bf16* p = pb + (wn * 32 + ni * 16 + l15) * LDK + kk + lhi * 8;
                v8bf lo = *(const v8bf*)p;
                v8bf hi = *(const v8bf*)(p + 16);
                bfr[ni] = __builtin_shufflevector(lo, hi, 0,1,2,3,4,5,6,7,8,9,10,11,12,13,14,15);
            }
            #pragma unroll
            for (int mi = 0; mi < 4; ++mi) {
                // A 16x32: lane<16 -> K {0..7,16..23}; lane>=16 -> K {8..15,24..31}
                const __bf16* p = pa + (wm * 64 + mi * 16 + l15) * LDK + kk + lhi * 8;
                v8bf lo = *(const v8bf*)p;
                v8bf hi = *(const v8bf*)(p + 16);
                af[mi] = __builtin_shufflevector(lo, hi, 0,1,2,3,4,5,6,7,8,9,10,11,12,13,14,15);
            }
            #pragma unroll
            for (int mi = 0; mi < 4; ++mi)
                #pragma unroll
                for (int ni = 0; ni < 2; ++ni)
                    acc[mi][ni] = __builtin_amdgcn_wmma_f32_16x16x32_bf16(
                        false, af[mi], false, bfr[ni], (short)0, acc[mi][ni], false, false);
        }

#if USE_ASYNC_LDS
        if (s + 1 < steps) wait_async0();   // stage s+1 landed in LDS (this wave's DMAs)
        __syncthreads();                    // all waves done computing cur + all DMAs visible
#else
        __syncthreads();
        if (s + 1 < steps) {
            const int nxt = 1 - cur;
            #pragma unroll
            for (int i = 0; i < 4; ++i) {
                *(uint4*)(&sA[nxt][rowS[i] * LDK + kcS[i]]) = pra[i];
                *(uint4*)(&sB[nxt][rowS[i] * LDK + kcS[i]]) = prb[i];
            }
            __syncthreads();
        }
#endif
    }

    // ---- epilogue; C/D layout: VGPR r, lanes 0-15 -> M=r, lanes 16-31 -> M=8+r; N=lane&15
    #pragma unroll
    for (int mi = 0; mi < 4; ++mi) {
        #pragma unroll
        for (int ni = 0; ni < 2; ++ni) {
            const long col = n0 + wn * 32 + ni * 16 + l15;
            const float bv = bias[col];
            #pragma unroll
            for (int r = 0; r < 8; ++r) {
                const long row = m0 + wm * 64 + mi * 16 + r + 8 * lhi;
                float v = acc[mi][ni][r] + bv;
                if (EPI == 0) {
                    v = v > 0.f ? v : 0.f;                     // ReLU
                    outB[row * (long)N + col] = (__bf16)v;
                } else {
                    v = (padmask[row] == 0) ? v : 0.f;         // zero padded tokens
                    outF[row * (long)N + col] = v;
                }
            }
        }
    }
}

// ---------------- launch ----------------

extern "C" void kernel_launch(void* const* d_in, const int* in_sizes, int n_in,
                              void* d_out, int out_size, void* d_ws, size_t ws_size,
                              hipStream_t stream)
{
    const float* x   = (const float*)d_in[0];
    const int*   pad = (const int*)  d_in[1];
    const float* W1  = (const float*)d_in[2];
    const float* b1  = (const float*)d_in[3];
    const float* W2  = (const float*)d_in[4];
    const float* b2  = (const float*)d_in[5];
    float* out = (float*)d_out;

    const int H = 1024, F = 4096;
    const long M = 4L * 4096L; // B * L

    __bf16* xb  = (__bf16*)d_ws;            // [M][H]
    __bf16* w1t = xb  + M * H;              // [F][H]
    __bf16* w2t = w1t + (long)F * H;        // [H][F]
    __bf16* hb  = w2t + (long)H * F;        // [M][F]

    // fp32 -> bf16 conversions / weight transposes (one-time, bandwidth-trivial)
    k_cvt_f32_to_bf16<<<(unsigned)((M * H) / 1024), 256, 0, stream>>>(x, xb, M * H);
    k_transpose_cvt<<<dim3(F / 32, H / 32), 256, 0, stream>>>(W1, w1t, H, F); // [H][F] -> [F][H]
    k_transpose_cvt<<<dim3(H / 32, F / 32), 256, 0, stream>>>(W2, w2t, F, H); // [F][H] -> [H][F]

    // GEMM1: h = relu(x*W1 + b1), bf16 out
    k_ffn_gemm<0><<<dim3(F / BN, (unsigned)(M / BM)), 256, 0, stream>>>(
        xb, w1t, b1, nullptr, nullptr, hb, (int)M, F, H);
    // GEMM2: out = mask .* (h*W2 + b2), fp32 out
    k_ffn_gemm<1><<<dim3(H / BN, (unsigned)(M / BM)), 256, 0, stream>>>(
        hb, w2t, b2, pad, out, nullptr, (int)M, H, F);
}
